// MambaLM_27204322853683
// MI455X (gfx1250) — matrix operand order repaired and checked by reading, not compiled, not run on hardware
//
#include <hip/hip_runtime.h>
#include <math.h>

#define Vv 32000
#define Dd 1024
#define NLAY 4
#define ED 2048
#define Nn 16
#define DR 64
#define DCV 4
#define Bb 2
#define Ll 1024
#define Mm (Bb*Ll)          // 2048 rows of activations
#define DBC (DR + 2*Nn)     // 96

typedef __attribute__((ext_vector_type(16))) __bf16 bf16x16;
typedef __attribute__((ext_vector_type(8)))  float  floatx8;

// ---------------- helpers ----------------
__device__ __forceinline__ unsigned short f2b(float f) {
  union { float f; unsigned u; } a; a.f = f;
  unsigned u = a.u;
  return (unsigned short)((u + 0x7FFFu + ((u >> 16) & 1u)) >> 16);  // RNE
}
__device__ __forceinline__ float siluf(float x) { return x / (1.f + __expf(-x)); }

// A-operand fragment: 16x32 bf16, per ISA 7.12.2 (lanes 0-15: K chunks 0..7 & 16..23;
// lanes 16-31: K chunks 8..15 & 24..31), row = M within lane group.
__device__ __forceinline__ bf16x16 load_fragA(const unsigned short* base, int ld,
                                              int row0, int k0, int lane) {
  int r = row0 + (lane & 15);
  const unsigned short* p = base + (size_t)r * ld + k0 + ((lane >> 4) << 3);
  union { uint4 u[2]; bf16x16 v; } f;
  f.u[0] = *(const uint4*)(p);
  f.u[1] = *(const uint4*)(p + 16);
  return f.v;
}
// B-operand fragment: 32x16 bf16 (K x N). W stored (N,K) row-major, so B[k,n]=W[n,k].
// Lanes 0-15 hold K=0..7 & 16..23, lanes 16-31 hold K=8..15 & 24..31; col = lane%16.
template<bool CLAMP>
__device__ __forceinline__ bf16x16 load_fragB(const unsigned short* base, int ld,
                                              int row0, int maxrow, int k0, int lane) {
  int r = row0 + (lane & 15);
  if (CLAMP) r = (r > maxrow) ? maxrow : r;
  const unsigned short* p = base + (size_t)r * ld + k0 + ((lane >> 4) << 3);
  union { uint4 u[2]; bf16x16 v; } f;
  f.u[0] = *(const uint4*)(p);
  f.u[1] = *(const uint4*)(p + 16);
  return f.v;
}

// ---------------- WMMA GEMM: C(MxN) = A(MxK) * W(NxK)^T ----------------
// EPI: 0 = plain store, 1 = +bias[n] then softplus, 2 = in-place residual add.
// EDGE: true only when Ntot is not a multiple of 128 (guards + clamped W rows).
template<int EPI, bool EDGE>
__global__ __launch_bounds__(128)
void gemm_bf16_wmma(const unsigned short* __restrict__ A, int lda,
                    const unsigned short* __restrict__ W, int ldw, int Nrows,
                    float* __restrict__ C, int ldc, int Ntot, int K,
                    const float* __restrict__ bias) {
  const int lane = threadIdx.x & 31;
  const int wv   = threadIdx.x >> 5;
  const int mw = blockIdx.y * 128 + (wv >> 1) * 64;
  const int nw = blockIdx.x * 128 + (wv & 1) * 64;

  floatx8 acc[4][4];
#pragma unroll
  for (int i = 0; i < 4; ++i)
#pragma unroll
    for (int j = 0; j < 4; ++j)
#pragma unroll
      for (int r = 0; r < 8; ++r) acc[i][j][r] = 0.f;

  for (int k = 0; k < K; k += 32) {
    bf16x16 af[4], bfr[4];
#pragma unroll
    for (int i = 0; i < 4; ++i)
      af[i] = load_fragA(A, lda, mw + 16 * i, k, lane);
#pragma unroll
    for (int j = 0; j < 4; ++j)
      bfr[j] = load_fragB<EDGE>(W, ldw, nw + 16 * j, Nrows - 1, k, lane);
#pragma unroll
    for (int i = 0; i < 4; ++i)
#pragma unroll
      for (int j = 0; j < 4; ++j)
        acc[i][j] = __builtin_amdgcn_wmma_f32_16x16x32_bf16(
            false, af[i], false, bfr[j], (short)0, acc[i][j], false, false);
  }

  const int moff = (lane >> 4) * 8;
  const int ncol = lane & 15;
#pragma unroll
  for (int i = 0; i < 4; ++i)
#pragma unroll
    for (int j = 0; j < 4; ++j) {
      const int n = nw + 16 * j + ncol;
      if (EDGE && n >= Ntot) continue;
      float bv = (EPI == 1) ? bias[n] : 0.f;
      float* cp = C + (size_t)(mw + 16 * i + moff) * ldc + n;
#pragma unroll
      for (int r = 0; r < 8; ++r) {
        float v = acc[i][j][r];
        if (EPI == 1) { v += bv; v = (v > 20.f) ? v : log1pf(__expf(v)); }
        if (EPI == 2) v += cp[(size_t)r * ldc];
        cp[(size_t)r * ldc] = v;
      }
    }
}

// ---------------- elementwise kernels ----------------
__global__ void f32_to_bf16_k(const float* __restrict__ in, unsigned short* __restrict__ out, size_t n) {
  size_t i = (size_t)blockIdx.x * blockDim.x + threadIdx.x;
  if (i < n) out[i] = f2b(in[i]);
}

__global__ void embed_k(const int* __restrict__ tok, const float* __restrict__ emb,
                        float* __restrict__ x) {
  size_t i = (size_t)blockIdx.x * blockDim.x + threadIdx.x;
  if (i >= (size_t)Mm * Dd) return;
  int m = (int)(i / Dd), d = (int)(i % Dd);
  x[i] = emb[(size_t)tok[m] * Dd + d];
}

__global__ void rmsnorm_bf16_k(const float* __restrict__ x, const float* __restrict__ w,
                               unsigned short* __restrict__ out) {
  __shared__ float red[256];
  int row = blockIdx.x;
  const float* xr = x + (size_t)row * Dd;
  float s = 0.f;
  for (int d = threadIdx.x; d < Dd; d += 256) { float v = xr[d]; s += v * v; }
  red[threadIdx.x] = s; __syncthreads();
  for (int off = 128; off > 0; off >>= 1) {
    if ((int)threadIdx.x < off) red[threadIdx.x] += red[threadIdx.x + off];
    __syncthreads();
  }
  float rs = rsqrtf(red[0] * (1.f / Dd) + 1e-5f);
  for (int d = threadIdx.x; d < Dd; d += 256)
    out[(size_t)row * Dd + d] = f2b(xr[d] * rs * w[d]);
}

// causal depthwise conv (DC=4) + bias + SiLU over xin = xz[:, :, :ED]
__global__ void conv_silu_k(const float* __restrict__ xz, const float* __restrict__ cw,
                            const float* __restrict__ cb, float* __restrict__ xs,
                            unsigned short* __restrict__ xsb) {
  size_t i = (size_t)blockIdx.x * blockDim.x + threadIdx.x;
  if (i >= (size_t)Mm * ED) return;
  int e = (int)(i % ED);
  int m = (int)(i / ED);
  int t = m % Ll;
  const float* w = cw + (size_t)e * DCV;
  float s = cb[e];
#pragma unroll
  for (int j = 0; j < DCV; ++j) {
    int tt = t - (DCV - 1) + j;
    if (tt >= 0) s += w[j] * xz[(size_t)(m - (DCV - 1) + j) * (2 * ED) + e];
  }
  float r = siluf(s);
  xs[i] = r;
  xsb[i] = f2b(r);
}

__global__ void extract_deltar_k(const float* __restrict__ dbc, unsigned short* __restrict__ drb) {
  size_t i = (size_t)blockIdx.x * blockDim.x + threadIdx.x;
  if (i >= (size_t)Mm * DR) return;
  int m = (int)(i / DR), c = (int)(i % DR);
  drb[i] = f2b(dbc[(size_t)m * DBC + c]);
}

// selective scan: one LANE per (b, e, n). 16 lanes (n=0..15) cooperate on one (b,e)
// channel; y = sum_n h[n]*C[n] via 4-step shfl_xor butterfly within 16-lane groups.
// Fuses +D*xs and the silu(z) gate; writes bf16 for the out_proj GEMM.
__global__ void scan_k(const float* __restrict__ delta, const float* __restrict__ dbc,
                       const float* __restrict__ xs, const float* __restrict__ xz,
                       const float* __restrict__ alog, const float* __restrict__ dpar,
                       unsigned short* __restrict__ yb) {
  int tid = blockIdx.x * blockDim.x + threadIdx.x;   // over Bb*ED*Nn = 65536
  int n = tid & (Nn - 1);
  int e = (tid >> 4) & (ED - 1);
  int b = tid >> 15;
  float A = -__expf(alog[(size_t)e * Nn + n]);
  float Dp = dpar[e];
  float h = 0.f;
  for (int t = 0; t < Ll; ++t) {
    size_t m = (size_t)b * Ll + t;
    float dt = delta[m * ED + e];          // broadcast across the 16 n-lanes
    float xv = xs[m * ED + e];             // broadcast
    float Bv = dbc[m * DBC + DR + n];      // coalesced across n
    float Cv = dbc[m * DBC + DR + Nn + n]; // coalesced across n
    h = __expf(dt * A) * h + dt * Bv * xv;
    float y = h * Cv;
    y += __shfl_xor(y, 8, 16);
    y += __shfl_xor(y, 4, 16);
    y += __shfl_xor(y, 2, 16);
    y += __shfl_xor(y, 1, 16);
    if (n == 0) {
      y += Dp * xv;
      float z = xz[m * (2 * ED) + ED + e];
      y *= siluf(z);
      yb[m * ED + e] = f2b(y);
    }
  }
}

// ---------------- launcher ----------------
extern "C" void kernel_launch(void* const* d_in, const int* in_sizes, int n_in,
                              void* d_out, int out_size, void* d_ws, size_t ws_size,
                              hipStream_t stream) {
  (void)in_sizes; (void)n_in; (void)out_size; (void)ws_size;
  const int*   tok  = (const int*)d_in[0];
  const float* emb  = (const float*)d_in[1];
  const float* ipw  = (const float*)d_in[2];
  const float* cw   = (const float*)d_in[3];
  const float* cb   = (const float*)d_in[4];
  const float* xpw  = (const float*)d_in[5];
  const float* dtw  = (const float*)d_in[6];
  const float* dtb  = (const float*)d_in[7];
  const float* alog = (const float*)d_in[8];
  const float* dpar = (const float*)d_in[9];
  const float* opw  = (const float*)d_in[10];
  const float* nw   = (const float*)d_in[11];
  const float* nfw  = (const float*)d_in[12];
  float* out = (float*)d_out;

  char* ws = (char*)d_ws;
  size_t off = 0;
  auto carve = [&](size_t bytes) -> char* {
    char* p = ws + off;
    off = (off + bytes + 255) & ~(size_t)255;
    return p;
  };
  unsigned short* emb_b = (unsigned short*)carve((size_t)Vv * Dd * 2);
  unsigned short* ipw_b = (unsigned short*)carve((size_t)NLAY * 2 * ED * Dd * 2);
  unsigned short* xpw_b = (unsigned short*)carve((size_t)NLAY * DBC * ED * 2);
  unsigned short* dtw_b = (unsigned short*)carve((size_t)NLAY * ED * DR * 2);
  unsigned short* opw_b = (unsigned short*)carve((size_t)NLAY * Dd * ED * 2);
  float*          x     = (float*)carve((size_t)Mm * Dd * 4);
  unsigned short* xnb   = (unsigned short*)carve((size_t)Mm * Dd * 2);
  float*          xzb   = (float*)carve((size_t)Mm * 2 * ED * 4);
  float*          xsf   = (float*)carve((size_t)Mm * ED * 4);
  unsigned short* xsb   = (unsigned short*)carve((size_t)Mm * ED * 2);
  float*          dbc   = (float*)carve((size_t)Mm * DBC * 4);
  unsigned short* drb   = (unsigned short*)carve((size_t)Mm * DR * 2);
  float*          delta = (float*)carve((size_t)Mm * ED * 4);
  unsigned short* yb    = (unsigned short*)carve((size_t)Mm * ED * 2);
  unsigned short* xfb   = (unsigned short*)carve((size_t)Mm * Dd * 2);

  size_t n;
  n = (size_t)Vv * Dd;            f32_to_bf16_k<<<(n + 255) / 256, 256, 0, stream>>>(emb, emb_b, n);
  n = (size_t)NLAY * 2 * ED * Dd; f32_to_bf16_k<<<(n + 255) / 256, 256, 0, stream>>>(ipw, ipw_b, n);
  n = (size_t)NLAY * DBC * ED;    f32_to_bf16_k<<<(n + 255) / 256, 256, 0, stream>>>(xpw, xpw_b, n);
  n = (size_t)NLAY * ED * DR;     f32_to_bf16_k<<<(n + 255) / 256, 256, 0, stream>>>(dtw, dtw_b, n);
  n = (size_t)NLAY * Dd * ED;     f32_to_bf16_k<<<(n + 255) / 256, 256, 0, stream>>>(opw, opw_b, n);

  n = (size_t)Mm * Dd;
  embed_k<<<(n + 255) / 256, 256, 0, stream>>>(tok, emb, x);

  const dim3 blk(128);
  for (int l = 0; l < NLAY; ++l) {
    rmsnorm_bf16_k<<<Mm, 256, 0, stream>>>(x, nw + (size_t)l * Dd, xnb);

    gemm_bf16_wmma<0, false><<<dim3((2 * ED) / 128, Mm / 128), blk, 0, stream>>>(
        xnb, Dd, ipw_b + (size_t)l * 2 * ED * Dd, Dd, 2 * ED,
        xzb, 2 * ED, 2 * ED, Dd, nullptr);

    n = (size_t)Mm * ED;
    conv_silu_k<<<(n + 255) / 256, 256, 0, stream>>>(
        xzb, cw + (size_t)l * ED * DCV, cb + (size_t)l * ED, xsf, xsb);

    gemm_bf16_wmma<0, true><<<dim3(1, Mm / 128), blk, 0, stream>>>(
        xsb, ED, xpw_b + (size_t)l * DBC * ED, ED, DBC,
        dbc, DBC, DBC, ED, nullptr);

    n = (size_t)Mm * DR;
    extract_deltar_k<<<(n + 255) / 256, 256, 0, stream>>>(dbc, drb);

    gemm_bf16_wmma<1, false><<<dim3(ED / 128, Mm / 128), blk, 0, stream>>>(
        drb, DR, dtw_b + (size_t)l * ED * DR, DR, ED,
        delta, ED, ED, DR, dtb + (size_t)l * ED);

    scan_k<<<(Bb * ED * Nn) / 256, 256, 0, stream>>>(
        delta, dbc, xsf, xzb, alog + (size_t)l * ED * Nn, dpar + (size_t)l * ED, yb);

    gemm_bf16_wmma<2, false><<<dim3(Dd / 128, Mm / 128), blk, 0, stream>>>(
        yb, ED, opw_b + (size_t)l * Dd * ED, ED, Dd,
        x, Dd, Dd, ED, nullptr);
  }

  rmsnorm_bf16_k<<<Mm, 256, 0, stream>>>(x, nfw, xfb);

  gemm_bf16_wmma<0, false><<<dim3(Vv / 128, Mm / 128), blk, 0, stream>>>(
      xfb, Dd, emb_b, Dd, Vv, out, Vv, Vv, Dd, nullptr);
}
